// VGNN_86397562127006
// MI455X (gfx1250) — matrix operand order repaired and checked by reading, not compiled, not run on hardware
//
#include <hip/hip_runtime.h>
#include <hip/hip_bf16.h>

// ---------------- model dimensions ----------------
#define BB    32            // batch
#define GGRID 14            // patches per side
#define NP    196           // patches per image
#define CC    320           // channel dim
#define KNN   9             // neighbors
#define LBLK  16            // ViG blocks
#define MROWS (BB * NP)     // 6272 rows
#define PATCH_F 768         // 3*16*16

typedef __attribute__((ext_vector_type(16))) __bf16 bf16x16;
typedef __attribute__((ext_vector_type(8)))  float  floatx8;

enum GemmFlags {
  F_BIAS = 1, F_AFFINE = 2, F_GELU_OUT = 4, F_GELU_A = 8, F_RESID = 16
};

__device__ __forceinline__ float vig_gelu(float x) {
  const float c = 0.7978845608028654f;                 // sqrt(2/pi)
  return 0.5f * x * (1.0f + tanhf(c * (x + 0.044715f * x * x * x)));
}

// ---- fragment helpers ------------------------------------------------------
// 16-bit fragment layout used here (16x16x32, contiguous k, not ISA table order):
//  A: lanes 0-15 row=lane, K=e               ; lanes 16-31: K += 16
//  B: lanes 0-15 col=lane, K=e               ; lanes 16-31: K += 16

__device__ __forceinline__ void loadA_fast(const float* __restrict__ p, float (&f)[16]) {
  float4 t0 = *(const float4*)(p +  0);
  float4 t1 = *(const float4*)(p +  4);
  float4 t2 = *(const float4*)(p +  8);
  float4 t3 = *(const float4*)(p + 12);
  f[0]=t0.x; f[1]=t0.y; f[2]=t0.z;  f[3]=t0.w;
  f[4]=t1.x; f[5]=t1.y; f[6]=t1.z;  f[7]=t1.w;
  f[8]=t2.x; f[9]=t2.y; f[10]=t2.z; f[11]=t2.w;
  f[12]=t3.x; f[13]=t3.y; f[14]=t3.z; f[15]=t3.w;
}

__device__ __forceinline__ void loadB_fast(const float* __restrict__ p, float (&f)[16]) {
  float4 t0 = *(const float4*)(p +  0);
  float4 t1 = *(const float4*)(p +  4);
  float4 t2 = *(const float4*)(p +  8);
  float4 t3 = *(const float4*)(p + 12);
  f[0]=t0.x; f[1]=t0.y; f[2]=t0.z;  f[3]=t0.w;
  f[4]=t1.x; f[5]=t1.y; f[6]=t1.z;  f[7]=t1.w;
  f[8]=t2.x; f[9]=t2.y; f[10]=t2.z; f[11]=t2.w;
  f[12]=t3.x; f[13]=t3.y; f[14]=t3.z; f[15]=t3.w;
}

// branchless tail loads: clamp address, select 0 (no EXEC divergence)
__device__ __forceinline__ void loadA_tail(const float* __restrict__ row,
                                           int k0, int ka0, int K, float (&f)[16]) {
  #pragma unroll
  for (int e = 0; e < 16; ++e) {
    int ka = k0 + ka0 + e;
    float v = row[(ka < K) ? ka : 0];
    f[e] = (ka < K) ? v : 0.0f;
  }
}

__device__ __forceinline__ void loadB_tail(const float* __restrict__ row,
                                           int k0, int kb0, int K, float (&f)[16]) {
  #pragma unroll
  for (int e = 0; e < 16; ++e) {
    int kb = k0 + kb0 + e;
    float v = row[(kb < K) ? kb : 0];
    f[e] = (kb < K) ? v : 0.0f;
  }
}

__device__ __forceinline__ void gelu16(float (&f)[16]) {
  #pragma unroll
  for (int e = 0; e < 16; ++e) f[e] = vig_gelu(f[e]);
}

__device__ __forceinline__ bf16x16 cvt16(const float (&f)[16]) {
  bf16x16 v;
  #pragma unroll
  for (int e = 0; e < 16; ++e) v[e] = (__bf16)f[e];
  return v;
}

// ------------------------------------------------------------------
// Fused GEMM: out[m,n] = epilogue( sum_k A[m,k] * W[n,k] )
// epilogue: (+bias[n]) -> (*scale[n]+shift[n]) -> gelu -> (+resid[m,n])
// A may be passed through gelu first (F_GELU_A).
// bf16 WMMA 16x16x32, fp32 accumulate.
// Wave tile 64x64 (4x4 fragments, 16 acc) -> 2 b128 loads per WMMA.
// Block = 4 waves stacked in M (256x64).  Batched via blockIdx.z.
// ------------------------------------------------------------------
__global__ __launch_bounds__(128)
void vig_gemm_bf16(const float* __restrict__ A, const float* __restrict__ W,
                   const float* __restrict__ bias, const float* __restrict__ scale,
                   const float* __restrict__ shift, const float* __restrict__ resid,
                   float* __restrict__ out, int M, int N, int K, int flags,
                   long long sA, long long sW, long long sO)
{
  A   += (long long)blockIdx.z * sA;
  W   += (long long)blockIdx.z * sW;
  out += (long long)blockIdx.z * sO;

  const int lane = threadIdx.x & 31;
  const int wave = threadIdx.x >> 5;
  const int row0 = (blockIdx.x * 4 + wave) * 64;
  const int col0 = blockIdx.y * 64;
  if (row0 >= M) return;                       // wave-uniform: EXEC all-1 at every WMMA

  const int hl = lane & 15;
  const int hi = lane >> 4;
  const int ka0 = hi * 16;
  const int kb0 = hi * 16;
  const bool geluA = (flags & F_GELU_A) != 0;

  const float* __restrict__ Arow[4];
  const float* __restrict__ Wrow[4];
  int ocol[4];
  #pragma unroll
  for (int mi = 0; mi < 4; ++mi) {
    int r = row0 + mi * 16 + hl;
    if (r >= M) r = M - 1;                     // dup row, never stored
    Arow[mi] = A + (long long)r * K;
  }
  #pragma unroll
  for (int ni = 0; ni < 4; ++ni) {
    int c = col0 + ni * 16 + hl;
    ocol[ni] = c;
    if (c >= N) c = N - 1;
    Wrow[ni] = W + (long long)c * K;
  }

  floatx8 acc[16] = {};

  const int kmain = K & ~31;
  for (int k0 = 0; k0 < kmain; k0 += 32) {
    bf16x16 a[4], b[4];
    #pragma unroll
    for (int mi = 0; mi < 4; ++mi) {
      float f[16];
      loadA_fast(Arow[mi] + k0 + ka0, f);
      if (geluA) gelu16(f);
      a[mi] = cvt16(f);
    }
    #pragma unroll
    for (int ni = 0; ni < 4; ++ni) {
      float f[16];
      loadB_fast(Wrow[ni] + k0 + kb0, f);
      b[ni] = cvt16(f);
    }
    if (k0 + 32 < K) {                         // CDNA5 global_prefetch of next K slice
      __builtin_prefetch(Arow[0] + k0 + 32, 0, 0);
      __builtin_prefetch(Wrow[0] + k0 + 32, 0, 0);
    }
    #pragma unroll
    for (int mi = 0; mi < 4; ++mi)
      #pragma unroll
      for (int ni = 0; ni < 4; ++ni)
        acc[mi * 4 + ni] = __builtin_amdgcn_wmma_f32_16x16x32_bf16(
            false, a[mi], false, b[ni], (short)0, acc[mi * 4 + ni], false, false);
  }

  if (K & 31) {                                // single branchless tail chunk
    const int k0 = kmain;
    bf16x16 a[4], b[4];
    #pragma unroll
    for (int mi = 0; mi < 4; ++mi) {
      float f[16];
      loadA_tail(Arow[mi], k0, ka0, K, f);
      if (geluA) gelu16(f);
      a[mi] = cvt16(f);
    }
    #pragma unroll
    for (int ni = 0; ni < 4; ++ni) {
      float f[16];
      loadB_tail(Wrow[ni], k0, kb0, K, f);
      b[ni] = cvt16(f);
    }
    #pragma unroll
    for (int mi = 0; mi < 4; ++mi)
      #pragma unroll
      for (int ni = 0; ni < 4; ++ni)
        acc[mi * 4 + ni] = __builtin_amdgcn_wmma_f32_16x16x32_bf16(
            false, a[mi], false, b[ni], (short)0, acc[mi * 4 + ni], false, false);
  }

  // epilogue — C/D layout: vgpr r -> row = r + 8*hi, col = hl
  #pragma unroll
  for (int ni = 0; ni < 4; ++ni) {
    const int c = ocol[ni];
    if (c < N) {
      float add = (flags & F_BIAS)   ? bias[c]  : 0.0f;
      float sc  = (flags & F_AFFINE) ? scale[c] : 1.0f;
      float sh  = (flags & F_AFFINE) ? shift[c] : 0.0f;
      #pragma unroll
      for (int mi = 0; mi < 4; ++mi) {
        #pragma unroll
        for (int r = 0; r < 8; ++r) {
          int orow = row0 + mi * 16 + hi * 8 + r;
          if (orow < M) {
            float v = acc[mi * 4 + ni][r] + add;
            if (flags & F_AFFINE)   v = v * sc + sh;
            if (flags & F_GELU_OUT) v = vig_gelu(v);
            long long o = (long long)orow * N + c;
            if (flags & F_RESID)    v += resid[o];
            out[o] = v;
          }
        }
      }
    }
  }
}

// ---------------- patchify: x[B,3,224,224] -> [B*N, 768] ----------------
__global__ void vig_patchify(const float* __restrict__ x, float* __restrict__ out) {
  int idx = blockIdx.x * blockDim.x + threadIdx.x;
  if (idx >= MROWS * PATCH_F) return;
  int f  = idx % PATCH_F;
  int bn = idx / PATCH_F;
  int b  = bn / NP, n = bn % NP;
  int gy = n / GGRID, gx = n % GGRID;
  int c  = f / 256, r = f % 256, py = r / 16, px = r % 16;
  out[idx] = x[(((long long)(b * 3 + c) * 224) + gy * 16 + py) * 224 + gx * 16 + px];
}

// ---------------- h += pos (broadcast over batch) ----------------
__global__ void vig_addpos(float* __restrict__ h, const float* __restrict__ pos) {
  int idx = blockIdx.x * blockDim.x + threadIdx.x;
  if (idx >= MROWS * CC) return;
  int c = idx % CC;
  int row = idx / CC;
  h[idx] += pos[(row % NP) * CC + c];
}

// ---------------- top-9 neighbor selection (lax.top_k tie rule) ----------------
__global__ void vig_topk(const float* __restrict__ sim, int* __restrict__ graph) {
  int idx = blockIdx.x * blockDim.x + threadIdx.x;     // b*NP + n
  if (idx >= MROWS) return;
  const float* row = sim + (long long)idx * NP;
  int chosen[KNN];
  #pragma unroll
  for (int k = 0; k < KNN; ++k) chosen[k] = -1;
  for (int k = 0; k < KNN; ++k) {
    float best = -3.4e38f; int bi = 0;
    for (int j = 0; j < NP; ++j) {
      bool taken = false;
      #pragma unroll
      for (int t = 0; t < KNN; ++t) taken |= (t < k) && (chosen[t] == j);
      float v = row[j];
      if (!taken && v > best) { best = v; bi = j; }   // strict > : lowest index wins ties
    }
    chosen[k] = bi;
    graph[(long long)idx * KNN + k] = bi;
  }
}

// ------- gather + max-relative + interleaved stack: st[.,2c]=u, st[.,2c+1]=maxrel -------
__global__ void vig_maxrel(const float* __restrict__ u, const int* __restrict__ graph,
                           float* __restrict__ st) {
  int idx = blockIdx.x * blockDim.x + threadIdx.x;     // over MROWS*CC
  if (idx >= MROWS * CC) return;
  int c  = idx % CC;
  int bn = idx / CC;
  int b  = bn / NP;
  float uc = u[(long long)bn * CC + c];
  const int* g = graph + (long long)bn * KNN;
  float m = -3.4e38f;
  #pragma unroll
  for (int k = 0; k < KNN; ++k) {
    float nv = u[((long long)b * NP + g[k]) * CC + c];
    m = fmaxf(m, nv - uc);
  }
  st[(long long)bn * (2 * CC) + 2 * c]     = uc;
  st[(long long)bn * (2 * CC) + 2 * c + 1] = m;
}

__global__ void vig_copy(const float* __restrict__ src, float* __restrict__ dst, int n) {
  int idx = blockIdx.x * blockDim.x + threadIdx.x;
  if (idx < n) dst[idx] = src[idx];
}

// ---------------- workspace layout (floats) ----------------
static const size_t SZ_WIDE  = (size_t)MROWS * 1280;
static const size_t SZ_H     = (size_t)MROWS * CC;
static const size_t OFF_WIDEA = 0;
static const size_t OFF_WIDEB = OFF_WIDEA + SZ_WIDE;
static const size_t OFF_HA    = OFF_WIDEB + SZ_WIDE;
static const size_t OFF_HB    = OFF_HA + SZ_H;
static const size_t OFF_U     = OFF_HB + SZ_H;
static const size_t OFF_Y     = OFF_U + SZ_H;
static const size_t OFF_SIM   = OFF_Y + SZ_H;
static const size_t OFF_GRAPH = OFF_SIM + (size_t)BB * NP * NP;   // int region

static inline void launch_gemm(const float* A, const float* W, const float* bias,
                               const float* s, const float* t, const float* resid,
                               float* out, int M, int N, int K, int flags,
                               int batch, long long sA, long long sW, long long sO,
                               hipStream_t stream) {
  dim3 grid((M + 255) / 256, (N + 63) / 64, batch);
  vig_gemm_bf16<<<grid, 128, 0, stream>>>(A, W, bias, s, t, resid, out,
                                          M, N, K, flags, sA, sW, sO);
}

extern "C" void kernel_launch(void* const* d_in, const int* in_sizes, int n_in,
                              void* d_out, int out_size, void* d_ws, size_t ws_size,
                              hipStream_t stream) {
  (void)in_sizes; (void)n_in; (void)out_size; (void)ws_size;

  const float* x   = (const float*)d_in[0];
  const float* pos = (const float*)d_in[25];

  float* ws    = (float*)d_ws;
  float* wideA = ws + OFF_WIDEA;
  float* wideB = ws + OFF_WIDEB;
  float* hA    = ws + OFF_HA;
  float* hB    = ws + OFF_HB;
  float* ubuf  = ws + OFF_U;
  float* ybuf  = ws + OFF_Y;
  float* sim   = ws + OFF_SIM;
  int*   graph = (int*)(ws + OFF_GRAPH);

  // ---- patchify ----
  {
    int total = MROWS * PATCH_F;
    vig_patchify<<<(total + 255) / 256, 256, 0, stream>>>(x, wideA);
  }

  // ---- patch embedding: 6 x (Linear -> affine -> gelu), last without gelu ----
  const int edims[7] = {768, 160, 80, 40, 80, 160, 320};
  const float* cur = wideA;
  for (int i = 0; i < 6; ++i) {
    const float* W = (const float*)d_in[1 + 4 * i];
    const float* b = (const float*)d_in[2 + 4 * i];
    const float* s = (const float*)d_in[3 + 4 * i];
    const float* t = (const float*)d_in[4 + 4 * i];
    float* o = (i == 5) ? hA : ((cur == wideA) ? wideB : wideA);
    int flags = F_BIAS | F_AFFINE | (i < 5 ? F_GELU_OUT : 0);
    launch_gemm(cur, W, b, s, t, nullptr, o, MROWS, edims[i + 1], edims[i],
                flags, 1, 0, 0, 0, stream);
    cur = o;
  }
  vig_addpos<<<(MROWS * CC + 255) / 256, 256, 0, stream>>>(hA, pos);

  // ---- 16 ViG blocks ----
  for (int l = 0; l < LBLK; ++l) {
    float* hcur  = (l & 1) ? hB : hA;
    float* hnext = (l & 1) ? hA : hB;

    auto P = [&](int idx, size_t stride) {
      return (const float*)d_in[idx] + (size_t)l * stride;
    };
    const float *i1W1 = P(26, 320 * 320), *i1b1 = P(27, 320),
                *i1W2 = P(28, 320 * 320), *i1b2 = P(29, 320),
                *i1s  = P(30, 320),       *i1t  = P(31, 320);
    const float *o1W1 = P(32, 320 * 320), *o1b1 = P(33, 320),
                *o1W2 = P(34, 320 * 320), *o1b2 = P(35, 320),
                *o1s  = P(36, 320),       *o1t  = P(37, 320);
    const float *i2W1 = P(38, 1280 * 320), *i2b1 = P(39, 1280),
                *i2W2 = P(40, 320 * 1280), *i2b2 = P(41, 320),
                *i2s  = P(42, 320),        *i2t  = P(43, 320);
    const float *o2W1 = P(44, 1280 * 320), *o2b1 = P(45, 1280),
                *o2W2 = P(46, 320 * 1280), *o2b2 = P(47, 320),
                *o2s  = P(48, 320),        *o2t  = P(49, 320);
    const float *fcW = P(50, 320 * 640), *fcb = P(51, 320);

    // sim[b] = h[b] @ h[b]^T  (batched WMMA GEMM, no bias)
    launch_gemm(hcur, hcur, nullptr, nullptr, nullptr, nullptr, sim,
                NP, NP, CC, 0, BB,
                (long long)NP * CC, (long long)NP * CC, (long long)NP * NP, stream);
    // top-9 neighbor indices
    vig_topk<<<(MROWS + 127) / 128, 128, 0, stream>>>(sim, graph);

    // u = two_layer(h, in1)
    launch_gemm(hcur, i1W1, i1b1, nullptr, nullptr, nullptr, ybuf,
                MROWS, CC, CC, F_BIAS | F_GELU_OUT, 1, 0, 0, 0, stream);
    launch_gemm(ybuf, i1W2, i1b2, i1s, i1t, nullptr, ubuf,
                MROWS, CC, CC, F_BIAS | F_AFFINE, 1, 0, 0, 0, stream);

    // st = interleave(u, maxrel)  -> wideB [MROWS, 640]
    vig_maxrel<<<(MROWS * CC + 255) / 256, 256, 0, stream>>>(ubuf, graph, wideB);

    // y = st @ fc_W^T + fc_b
    launch_gemm(wideB, fcW, fcb, nullptr, nullptr, nullptr, ybuf,
                MROWS, CC, 2 * CC, F_BIAS, 1, 0, 0, 0, stream);

    // v = two_layer(gelu(y), out1) + h
    launch_gemm(ybuf, o1W1, o1b1, nullptr, nullptr, nullptr, ubuf,
                MROWS, CC, CC, F_BIAS | F_GELU_A | F_GELU_OUT, 1, 0, 0, 0, stream);
    launch_gemm(ubuf, o1W2, o1b2, o1s, o1t, hcur, hnext,
                MROWS, CC, CC, F_BIAS | F_AFFINE | F_RESID, 1, 0, 0, 0, stream);

    // t1 = two_layer(v, in2)
    launch_gemm(hnext, i2W1, i2b1, nullptr, nullptr, nullptr, wideA,
                MROWS, 4 * CC, CC, F_BIAS | F_GELU_OUT, 1, 0, 0, 0, stream);
    launch_gemm(wideA, i2W2, i2b2, i2s, i2t, nullptr, ybuf,
                MROWS, CC, 4 * CC, F_BIAS | F_AFFINE, 1, 0, 0, 0, stream);

    // h_next = two_layer(gelu(t1), out2) + v   (residual in-place on hnext)
    launch_gemm(ybuf, o2W1, o2b1, nullptr, nullptr, nullptr, wideA,
                MROWS, 4 * CC, CC, F_BIAS | F_GELU_A | F_GELU_OUT, 1, 0, 0, 0, stream);
    launch_gemm(wideA, o2W2, o2b2, o2s, o2t, hnext, hnext,
                MROWS, CC, 4 * CC, F_BIAS | F_AFFINE | F_RESID, 1, 0, 0, 0, stream);
  }

  // after an even number of blocks the result is in hA
  vig_copy<<<(MROWS * CC + 255) / 256, 256, 0, stream>>>(hA, (float*)d_out, MROWS * CC);
}
